// GNNTemporalFusionTransformer_21337397527221
// MI455X (gfx1250) — matrix operand order repaired and checked by reading, not compiled, not run on hardware
//
#include <hip/hip_runtime.h>
#include <hip/hip_bf16.h>

// ---------------------------------------------------------------------------
// GNN Temporal Fusion Transformer for MI455X (gfx1250, wave32, WMMA + TDM)
//   - all dense GEMMs on v_wmma_f32_16x16x32_f16 (f16 in, f32 acc)
//   - VSN per-feature GRN fused, weights streamed by the Tensor Data Mover
//     with a double-buffered TENSORcnt pipeline (hides global latency)
//   - LSTM recurrence fused; Whh^T staged to LDS once per WG via TDM
// ---------------------------------------------------------------------------

#define BQ   2
#define LQ   24
#define NQ_  256
#define FQ   8
#define DQ   64
#define GHQ  2
#define EQ   1024
#define EDQ  4
#define NSQ  8
#define HZQ  12
#define NTQ  2
#define NQQ  3
#define GLQ  (BQ * LQ)      /* 48  graphs              */
#define MSEQ (BQ * NQ_)     /* 512 sequences           */
#define MLQ  (MSEQ * LQ)    /* 12288 rows              */
#define ETOT (EQ + NQ_)     /* 1280 edges (w/ selfloop)*/
#define OUT0 (BQ * HZQ * NQ_ * NTQ * NQQ) /* 36864 */

typedef __attribute__((ext_vector_type(16))) _Float16 v16h;
typedef __attribute__((ext_vector_type(8)))  _Float16 v8h;
typedef __attribute__((ext_vector_type(8)))  float    v8f;

// ---------------- Tensor Data Mover (gfx1250 TDM) --------------------------
#if defined(__has_builtin)
#  if __has_builtin(__builtin_amdgcn_tensor_load_to_lds)
#    define HAVE_TDM 1
#  endif
#endif

// Compile-probe marker: tells us (via stderr) which weight-staging path is live.
#ifdef HAVE_TDM
#warning "CDNA5-PROBE: TDM path ACTIVE (tensor_load_to_lds emitted)"
#else
#warning "CDNA5-PROBE: TDM builtin NOT found, cooperative-copy fallback in use"
#endif

typedef __attribute__((ext_vector_type(4))) unsigned tdm_u32x4;
typedef __attribute__((ext_vector_type(8))) int      tdm_i32x8;
typedef __attribute__((ext_vector_type(4))) int      tdm_i32x4;

#ifdef HAVE_TDM
#define TDM_WAIT(n) __builtin_amdgcn_s_wait_tensorcnt((short)(n))
#else
#define TDM_WAIT(n)
#endif

// Issue one 2-D f16 tile copy global->LDS via TDM (D# per ISA 8.3/8.4).
// rows x cols elements, contiguous in both global and LDS.
__device__ __forceinline__ void tdm_load_2d_f16(const _Float16* gsrc, void* lds_dst,
                                                unsigned rows, unsigned cols) {
#ifdef HAVE_TDM
  unsigned long long ga = (unsigned long long)(size_t)gsrc;
  unsigned lo = (unsigned)(size_t)lds_dst;        // LDS byte offset (flat low bits)
  tdm_u32x4 g0;
  g0[0] = 1u;                                     // count=1 valid, user descriptor
  g0[1] = lo;                                     // lds_addr
  g0[2] = (unsigned)(ga & 0xffffffffu);           // global_addr[31:0]
  g0[3] = (unsigned)((ga >> 32) & 0x01ffffffu) | 0x80000000u; // addr[56:32] | type=2
  tdm_i32x8 g1;
  g1[0] = (int)(1u << 16);                        // data_size=1 (2B), no multicast
  g1[1] = (int)(cols << 16);                      // tensor_dim0[15:0] @ bits 63:48
  g1[2] = (int)((cols >> 16) | (rows << 16));     // dim0[31:16] | dim1[15:0]
  g1[3] = (int)((rows >> 16) | (cols << 16));     // dim1[31:16] | tile_dim0
  g1[4] = (int)(rows & 0xffffu);                  // tile_dim1 | tile_dim2=0
  g1[5] = (int)cols;                              // tensor_dim0_stride[31:0]
  g1[6] = 0;                                      // stride0[47:32] | stride1[15:0]
  g1[7] = 0;                                      // stride1[47:16]
  tdm_i32x4 z4 = {0, 0, 0, 0};
#if __clang_major__ >= 23
  tdm_i32x8 z8 = {0, 0, 0, 0, 0, 0, 0, 0};
  __builtin_amdgcn_tensor_load_to_lds(g0, g1, z4, z4, z8, 0);
#else
  __builtin_amdgcn_tensor_load_to_lds(g0, g1, z4, z4, 0);
#endif
#else
  (void)gsrc; (void)lds_dst; (void)rows; (void)cols;
#endif
}

// ---- WMMA fragment loaders (ISA 7.12.2 layouts) ---------------------------
__device__ __forceinline__ v16h load_frag_a(const _Float16* base, int stride, int lane) {
  int r  = lane & 15;
  int kb = (lane >> 4) << 3;
  const _Float16* p = base + r * stride + kb;
  v8h lo = *(const v8h*)(p);
  v8h hi = *(const v8h*)(p + 16);
  v16h o;
#pragma unroll
  for (int i = 0; i < 8; ++i) { o[i] = lo[i]; o[i + 8] = hi[i]; }
  return o;
}
__device__ __forceinline__ v16h load_frag_b(const _Float16* base, int stride, int lane) {
  int c  = lane & 15;
  int kb = (lane >> 4) << 3;
  const _Float16* p = base + c * stride + kb;
  v8h lo = *(const v8h*)(p);
  v8h hi = *(const v8h*)(p + 16);
  v16h o;
#pragma unroll
  for (int i = 0; i < 8; ++i) { o[i] = lo[i]; o[i + 8] = hi[i]; }
  return o;
}

__device__ __forceinline__ float sigf(float x) { return 1.0f / (1.0f + __expf(-x)); }
__device__ __forceinline__ unsigned omap(float f) {
  unsigned u = __float_as_uint(f);
  return (u & 0x80000000u) ? ~u : (u | 0x80000000u);
}
__device__ __forceinline__ float ounmap(unsigned u) {
  return __uint_as_float((u & 0x80000000u) ? (u & 0x7fffffffu) : ~u);
}

// ---- generic WMMA GEMM: C[M,N] = act(A[M,K]@W + bias), Bt = W^T as [N][K] --
__global__ __launch_bounds__(128) void gemm_wmma(
    const _Float16* __restrict__ A, int lda,
    const _Float16* __restrict__ Bt, int ldb,
    const float* __restrict__ bias,
    float* __restrict__ C, int ldc,
    _Float16* __restrict__ Ch, int ldch,
    int N, int Kpad, int act) {
  int m0   = blockIdx.x * 16;
  int wave = threadIdx.x >> 5;
  int lane = threadIdx.x & 31;
  int n0   = blockIdx.y * 64 + wave * 16;
  v8f acc = {};
  const _Float16* arow = A + (size_t)m0 * lda;
  const _Float16* brow = Bt + (size_t)n0 * ldb;
  for (int k0 = 0; k0 < Kpad; k0 += 32) {
    __builtin_prefetch(arow + k0 + 32, 0, 3);  // global_prefetch_b8, WGP scope
    v16h fa = load_frag_a(arow + k0, lda, lane);
    v16h fb = load_frag_b(brow + k0, ldb, lane);
    acc = __builtin_amdgcn_wmma_f32_16x16x32_f16(false, fa, false, fb, (short)0, acc, false, false);
  }
  int col = n0 + (lane & 15);
  int r0  = (lane >> 4) * 8;
  if (col < N) {
    float bv = bias ? bias[col] : 0.0f;
#pragma unroll
    for (int i = 0; i < 8; ++i) {
      float v = acc[i] + bv;
      if (act == 1) v = fmaxf(v, 0.0f);
      else if (act == 2) v = sigf(v);
      C[(size_t)(m0 + r0 + i) * ldc + col] = v;
      if (Ch) Ch[(size_t)(m0 + r0 + i) * ldch + col] = (_Float16)v;
    }
  }
}

// ---- weight prep: W[K][N] f32 -> Bt[Npad][Kpad] f16 (zero padded) ---------
__global__ void prep_bt(const float* __restrict__ W, _Float16* __restrict__ Bt,
                        int K, int N, int Kpad, int Npad) {
  int idx = blockIdx.x * blockDim.x + threadIdx.x;
  if (idx >= Npad * Kpad) return;
  int n = idx / Kpad, k = idx % Kpad;
  Bt[idx] = (_Float16)((n < N && k < K) ? W[k * N + n] : 0.0f);
}
__global__ void prep_bt3(const float* __restrict__ W, _Float16* __restrict__ Bt) {
  int idx = blockIdx.x * blockDim.x + threadIdx.x;
  if (idx >= 64 * 64 * 64) return;
  int f = idx >> 12, e = (idx >> 6) & 63, d = idx & 63;
  Bt[idx] = (_Float16)W[(f * 64 + d) * 64 + e];
}

__global__ void fill_u32(unsigned* p, unsigned v, int n) {
  int i = blockIdx.x * blockDim.x + threadIdx.x; if (i < n) p[i] = v;
}
__global__ void fill_f32(float* p, float v, int n) {
  int i = blockIdx.x * blockDim.x + threadIdx.x; if (i < n) p[i] = v;
}
__global__ void to_f16(const float* __restrict__ x, _Float16* __restrict__ y, int n) {
  int i = blockIdx.x * blockDim.x + threadIdx.x; if (i < n) y[i] = (_Float16)x[i];
}

__global__ void static_emb(const float* __restrict__ nf, const float* __restrict__ W,
                           const float* __restrict__ b, float* __restrict__ out) {
  int idx = blockIdx.x * blockDim.x + threadIdx.x;
  if (idx >= NQ_ * DQ) return;
  int n = idx >> 6, d = idx & 63;
  float s = b[d];
  for (int k = 0; k < NSQ; ++k) s += nf[n * NSQ + k] * W[k * DQ + d];
  out[idx] = fmaxf(s, 0.0f);
}

__global__ void build_node_in(const float* __restrict__ x, const float* __restrict__ stat,
                              _Float16* __restrict__ out) {
  int idx = blockIdx.x * blockDim.x + threadIdx.x;
  if (idx >= MLQ * 96) return;
  int r = idx / 96, k = idx % 96;
  float v = 0.0f;
  if (k < FQ) v = x[(size_t)r * FQ + k];
  else if (k < FQ + DQ) v = stat[(r % NQ_) * DQ + (k - FQ)];
  out[idx] = (_Float16)v;
}

__global__ void ea_mean(const float* __restrict__ ea, float* __restrict__ mean) {
  int t = threadIdx.x;
  if (t < EDQ) {
    float s = 0.0f;
    for (int e = 0; e < EQ; ++e) s += ea[e * EDQ + t];
    mean[t] = s / (float)EQ;
  }
}
__global__ void build_edges(const int* __restrict__ ei, const float* __restrict__ ea,
                            const float* __restrict__ mean, int* __restrict__ src,
                            int* __restrict__ tgt, float* __restrict__ ea_ext) {
  int i = blockIdx.x * blockDim.x + threadIdx.x;
  if (i >= ETOT) return;
  if (i < EQ) {
    src[i] = ei[i]; tgt[i] = ei[EQ + i];
    for (int s = 0; s < EDQ; ++s) ea_ext[i * EDQ + s] = ea[i * EDQ + s];
  } else {
    src[i] = tgt[i] = i - EQ;
    for (int s = 0; s < EDQ; ++s) ea_ext[i * EDQ + s] = mean[s];
  }
}
__global__ void e_proj(const float* __restrict__ ea, const float* __restrict__ We,
                       float* __restrict__ e) {
  int idx = blockIdx.x * blockDim.x + threadIdx.x;
  if (idx >= ETOT * GHQ * DQ) return;
  int i = idx / 128, j = idx % 128;
  float s = 0.0f;
  for (int k = 0; k < EDQ; ++k) s += ea[i * EDQ + k] * We[k * 128 + j];
  e[idx] = s;
}

// ---- GATv2 edge passes ----------------------------------------------------
__global__ void gat_pass1(const float* __restrict__ xl, const float* __restrict__ xr,
                          const float* __restrict__ e, const float* __restrict__ att,
                          const int* __restrict__ src, const int* __restrict__ tgt,
                          float* __restrict__ abuf, unsigned* __restrict__ amax) {
  int idx = blockIdx.x * blockDim.x + threadIdx.x;
  if (idx >= GLQ * ETOT * GHQ) return;
  int h = idx & 1, ei = (idx >> 1) % ETOT, g = idx / (ETOT * GHQ);
  int s = src[ei], t = tgt[ei];
  const float* pl = xl + ((size_t)(g * NQ_ + s)) * 128 + h * 64;
  const float* pr = xr + ((size_t)(g * NQ_ + t)) * 128 + h * 64;
  const float* pe = e + (size_t)ei * 128 + h * 64;
  const float* pa = att + h * 64;
  float a = 0.0f;
  for (int d = 0; d < 64; ++d) {
    float m = pl[d] + pr[d] + pe[d];
    m = (m > 0.0f) ? m : 0.2f * m;
    a += m * pa[d];
  }
  abuf[idx] = a;
  atomicMax(&amax[(g * NQ_ + t) * GHQ + h], omap(a));
}
__global__ void gat_pass2(const float* __restrict__ abuf, const unsigned* __restrict__ amax,
                          const int* __restrict__ tgt, float* __restrict__ exbuf,
                          float* __restrict__ den) {
  int idx = blockIdx.x * blockDim.x + threadIdx.x;
  if (idx >= GLQ * ETOT * GHQ) return;
  int h = idx & 1, ei = (idx >> 1) % ETOT, g = idx / (ETOT * GHQ);
  int t = tgt[ei];
  float ex = __expf(abuf[idx] - ounmap(amax[(g * NQ_ + t) * GHQ + h]));
  exbuf[idx] = ex;
  atomicAdd(&den[(g * NQ_ + t) * GHQ + h], ex);
}
__global__ void gat_pass3(const float* __restrict__ exbuf, const float* __restrict__ den,
                          const float* __restrict__ xl, const int* __restrict__ src,
                          const int* __restrict__ tgt, float* __restrict__ gout) {
  int idx = blockIdx.x * blockDim.x + threadIdx.x;
  if (idx >= GLQ * ETOT * GHQ * 64) return;
  int d = idx & 63, r = idx >> 6;
  int h = r & 1, ei = (r >> 1) % ETOT, g = r / (ETOT * GHQ);
  int s = src[ei], t = tgt[ei];
  float alpha = exbuf[r] / den[(g * NQ_ + t) * GHQ + h];
  float xj = xl[((size_t)(g * NQ_ + s)) * 128 + h * 64 + d];
  atomicAdd(&gout[((size_t)(g * NQ_ + t) * GHQ + h) * 64 + d], alpha * xj);
}
__global__ void gat_mean(const float* __restrict__ gout, const float* __restrict__ b,
                         float* __restrict__ out, _Float16* __restrict__ outh) {
  int idx = blockIdx.x * blockDim.x + threadIdx.x;
  if (idx >= MLQ * 64) return;
  int d = idx & 63; size_t rn = idx >> 6;
  float v = 0.5f * (gout[(rn * 2 + 0) * 64 + d] + gout[(rn * 2 + 1) * 64 + d]) + b[d];
  v = fmaxf(v, 0.0f);
  out[idx] = v; outh[idx] = (_Float16)v;
}

__global__ void transpose_flat(const float* __restrict__ h, float* __restrict__ out,
                               _Float16* __restrict__ outh) {
  int idx = blockIdx.x * blockDim.x + threadIdx.x;
  if (idx >= MLQ * 64) return;
  int d = idx & 63, r = idx >> 6;
  int l = r % LQ, bn = r / LQ, n = bn % NQ_, b = bn / NQ_;
  float v = h[((size_t)((b * LQ + l) * NQ_ + n)) * 64 + d];
  out[idx] = v; outh[idx] = (_Float16)v;
}

// ---- GRN tail: LN(x + hf*hg) (or LN(x+hf) if hg null), optional softmax ---
__global__ __launch_bounds__(64) void grn_finish(
    const float* __restrict__ x, const float* __restrict__ hf, const float* __restrict__ hg,
    const float* __restrict__ g, const float* __restrict__ be,
    float* __restrict__ out, _Float16* __restrict__ outh, int softmax_last) {
  __shared__ float red[64];
  int row = blockIdx.x, d = threadIdx.x;
  size_t o = (size_t)row * 64 + d;
  float v = x[o] + (hg ? hf[o] * hg[o] : hf[o]);
  red[d] = v; __syncthreads();
  for (int s = 32; s > 0; s >>= 1) { if (d < s) red[d] += red[d + s]; __syncthreads(); }
  float m = red[0] / 64.0f; __syncthreads();
  float c = v - m;
  red[d] = c * c; __syncthreads();
  for (int s = 32; s > 0; s >>= 1) { if (d < s) red[d] += red[d + s]; __syncthreads(); }
  float var = red[0] / 64.0f; __syncthreads();
  float y = c * rsqrtf(var + 1e-5f) * g[d] + be[d];
  if (softmax_last) {
    red[d] = y; __syncthreads();
    for (int s = 32; s > 0; s >>= 1) { if (d < s) red[d] = fmaxf(red[d], red[d + s]); __syncthreads(); }
    float mx = red[0]; __syncthreads();
    float ex = __expf(y - mx);
    red[d] = ex; __syncthreads();
    for (int s = 32; s > 0; s >>= 1) { if (d < s) red[d] += red[d + s]; __syncthreads(); }
    y = ex / red[0]; __syncthreads();
  }
  out[o] = y;
  if (outh) outh[o] = (_Float16)y;
}

// ---- FUSED per-feature VSN GRN: TDM double-buffered weight stream ---------
__global__ __launch_bounds__(128) void vsn_per(
    const float* __restrict__ xf, const float* __restrict__ w,
    const float* __restrict__ W1, const float* __restrict__ b1,
    const _Float16* __restrict__ W2t, const float* __restrict__ b2,
    const _Float16* __restrict__ Wft, const float* __restrict__ bfp,
    const _Float16* __restrict__ Wgt, const float* __restrict__ bgp,
    const float* __restrict__ Ws, const float* __restrict__ bs,
    const float* __restrict__ g, const float* __restrict__ be,
    float* __restrict__ vout, _Float16* __restrict__ vouth) {
  __shared__ _Float16 h1[16][72];
  __shared__ _Float16 h2[16][72];
  __shared__ float pr[16][68];
  __shared__ float acc[16][64];
  __shared__ float xv[16][64];
  __shared__ float wv[16][64];
  __shared__ float mrow[16], rrow[16];
  __shared__ __attribute__((aligned(128))) _Float16 wbuf[2][3][64][64]; // 48KB, TDM dest
  int m0 = blockIdx.x * 16;
  int tid = threadIdx.x, wave = tid >> 5, lane = tid & 31;
  int rr = tid >> 3, cb = (tid & 7) * 8;

  auto stage = [&](int f, int b) {
#ifdef HAVE_TDM
    if (tid < 32) {  // one wave issues the three async tile descriptors
      tdm_load_2d_f16(W2t + (size_t)f * 4096, &wbuf[b][0][0][0], 64, 64);
      tdm_load_2d_f16(Wft + (size_t)f * 4096, &wbuf[b][1][0][0], 64, 64);
      tdm_load_2d_f16(Wgt + (size_t)f * 4096, &wbuf[b][2][0][0], 64, 64);
    }
#else
    for (int i = tid; i < 512; i += 128) {
      ((v8h*)&wbuf[b][0][0][0])[i] = ((const v8h*)(W2t + (size_t)f * 4096))[i];
      ((v8h*)&wbuf[b][1][0][0])[i] = ((const v8h*)(Wft + (size_t)f * 4096))[i];
      ((v8h*)&wbuf[b][2][0][0])[i] = ((const v8h*)(Wgt + (size_t)f * 4096))[i];
    }
#endif
  };

#pragma unroll
  for (int j = 0; j < 8; ++j) {
    xv[rr][cb + j] = xf[(size_t)(m0 + rr) * 64 + cb + j];
    wv[rr][cb + j] = w[(size_t)(m0 + rr) * 64 + cb + j];
    acc[rr][cb + j] = 0.0f;
  }
#ifdef HAVE_TDM
  stage(0, 0);                 // prime the pipeline (async)
#endif
  __syncthreads();

  for (int f = 0; f < 64; ++f) {
    int b = f & 1;
#ifdef HAVE_TDM
    if (f < 63) stage(f + 1, 1 - b);       // issue next feature's panels (async)
    if (tid < 32) {
      if (f < 63) TDM_WAIT(3);             // current 3 panels done, next 3 in flight
      else        TDM_WAIT(0);
    }
    __syncthreads();                       // wbuf[b] visible to all waves
#else
    stage(f, b);
    __syncthreads();
#endif
    // h1 = relu(x[:,f]*W1[f] + b1[f])
#pragma unroll
    for (int j = 0; j < 8; ++j) {
      int d = cb + j;
      h1[rr][d] = (_Float16)fmaxf(xv[rr][f] * W1[f * 64 + d] + b1[f * 64 + d], 0.0f);
    }
    __syncthreads();
    // h2 = h1 @ W2[f] + b2[f]
    {
      int cw = wave * 16;
      v8f a2 = {};
      for (int k0 = 0; k0 < 64; k0 += 32) {
        v16h fa = load_frag_a(&h1[0][k0], 72, lane);
        v16h fb = load_frag_b(&wbuf[b][0][cw][0] + k0, 64, lane);
        a2 = __builtin_amdgcn_wmma_f32_16x16x32_f16(false, fa, false, fb, (short)0, a2, false, false);
      }
      int col = cw + (lane & 15), r0 = (lane >> 4) * 8;
      float bb = b2[f * 64 + col];
#pragma unroll
      for (int i = 0; i < 8; ++i) h2[r0 + i][col] = (_Float16)(a2[i] + bb);
    }
    __syncthreads();
    // pr = (h2@Wf+bf)*sigmoid(h2@Wg+bg) + (x*Ws+bs)
    {
      int cw = wave * 16;
      v8f af = {}, ag = {};
      for (int k0 = 0; k0 < 64; k0 += 32) {
        v16h fa  = load_frag_a(&h2[0][k0], 72, lane);
        v16h fbf = load_frag_b(&wbuf[b][1][cw][0] + k0, 64, lane);
        v16h fbg = load_frag_b(&wbuf[b][2][cw][0] + k0, 64, lane);
        af = __builtin_amdgcn_wmma_f32_16x16x32_f16(false, fa, false, fbf, (short)0, af, false, false);
        ag = __builtin_amdgcn_wmma_f32_16x16x32_f16(false, fa, false, fbg, (short)0, ag, false, false);
      }
      int col = cw + (lane & 15), r0 = (lane >> 4) * 8;
      float bF = bfp[f * 64 + col], bG = bgp[f * 64 + col];
      float ws_ = Ws[f * 64 + col], bs_ = bs[f * 64 + col];
#pragma unroll
      for (int i = 0; i < 8; ++i) {
        float hv = (af[i] + bF) * sigf(ag[i] + bG);
        pr[r0 + i][col] = hv + xv[r0 + i][f] * ws_ + bs_;
      }
    }
    __syncthreads();
    if (tid < 16) {
      float s = 0.0f;
      for (int d = 0; d < 64; ++d) s += pr[tid][d];
      float m = s / 64.0f, v2 = 0.0f;
      for (int d = 0; d < 64; ++d) { float c = pr[tid][d] - m; v2 += c * c; }
      mrow[tid] = m; rrow[tid] = rsqrtf(v2 / 64.0f + 1e-5f);
    }
    __syncthreads();
#pragma unroll
    for (int j = 0; j < 8; ++j) {
      int d = cb + j;
      float y = (pr[rr][d] - mrow[rr]) * rrow[rr] * g[f * 64 + d] + be[f * 64 + d];
      acc[rr][d] += wv[rr][f] * y;
    }
    __syncthreads();
  }
#pragma unroll
  for (int j = 0; j < 8; ++j) {
    int d = cb + j;
    float v = acc[rr][d];
    vout[(size_t)(m0 + rr) * 64 + d] = v;
    vouth[(size_t)(m0 + rr) * 64 + d] = (_Float16)v;
  }
}

// ---- fused LSTM layer: Whh^T staged to LDS by TDM, WMMA per step ----------
__global__ __launch_bounds__(128) void lstm_layer(
    const float* __restrict__ xW,      // [MLQ][256], row = seq*24+t
    const _Float16* __restrict__ Whht, // [256][64] f16
    float* __restrict__ hout, _Float16* __restrict__ houth) {
  __shared__ float hS[16][64], cS[16][64];
  __shared__ _Float16 hh[16][72];
  __shared__ float gates[16][256];
  __shared__ __attribute__((aligned(128))) _Float16 wlds[256][64]; // 32KB, TDM dest
  int m0 = blockIdx.x * 16;
  int tid = threadIdx.x, wave = tid >> 5, lane = tid & 31;
  int rr = tid >> 3, cb = (tid & 7) * 8;
#ifdef HAVE_TDM
  if (tid < 32) {
    tdm_load_2d_f16(Whht, &wlds[0][0], 256, 64);  // one async DMA of 32KB
    TDM_WAIT(0);
  }
#else
  for (int i = tid; i < 2048; i += 128)
    ((v8h*)&wlds[0][0])[i] = ((const v8h*)Whht)[i];
#endif
#pragma unroll
  for (int j = 0; j < 8; ++j) { hS[rr][cb + j] = 0.0f; cS[rr][cb + j] = 0.0f; }
  __syncthreads();
  for (int t = 0; t < LQ; ++t) {
#pragma unroll
    for (int j = 0; j < 8; ++j) hh[rr][cb + j] = (_Float16)hS[rr][cb + j];
    __syncthreads();
    for (int ct = 0; ct < 4; ++ct) {
      int col0 = wave * 64 + ct * 16;
      v8f a = {};
      for (int k0 = 0; k0 < 64; k0 += 32) {
        v16h fa = load_frag_a(&hh[0][k0], 72, lane);
        v16h fb = load_frag_b(&wlds[col0][0] + k0, 64, lane);
        a = __builtin_amdgcn_wmma_f32_16x16x32_f16(false, fa, false, fb, (short)0, a, false, false);
      }
      int col = col0 + (lane & 15), r0 = (lane >> 4) * 8;
#pragma unroll
      for (int i = 0; i < 8; ++i)
        gates[r0 + i][col] = a[i] + xW[((size_t)(m0 + r0 + i) * LQ + t) * 256 + col];
    }
    __syncthreads();
#pragma unroll
    for (int j = 0; j < 8; ++j) {
      int d = cb + j;
      float gi = gates[rr][d], gf = gates[rr][64 + d], gg = gates[rr][128 + d], go = gates[rr][192 + d];
      float cn = sigf(gf) * cS[rr][d] + sigf(gi) * tanhf(gg);
      float hn = sigf(go) * tanhf(cn);
      cS[rr][d] = cn; hS[rr][d] = hn;
      size_t o = ((size_t)(m0 + rr) * LQ + t) * 64 + d;
      hout[o] = hn; houth[o] = (_Float16)hn;
    }
    __syncthreads();
  }
}

// ---- MHA core (24x24 per head, tiny; VALU) --------------------------------
__global__ __launch_bounds__(128) void mha_attn(const float* __restrict__ qkv,
                                                float* __restrict__ o) {
  __shared__ float q[24][64], k[24][64], v[24][64];
  int m = blockIdx.x, tid = threadIdx.x;
  for (int i = tid; i < 24 * 64; i += 128) {
    int l = i >> 6, d = i & 63;
    const float* row = qkv + ((size_t)m * LQ + l) * 192;
    q[l][d] = row[d]; k[l][d] = row[64 + d]; v[l][d] = row[128 + d];
  }
  __syncthreads();
  if (tid < 96) {
    int l = tid >> 2, h = tid & 3;
    float sc[24], mx = -1e30f;
    for (int s = 0; s < 24; ++s) {
      float a = 0.0f;
      for (int d = 0; d < 16; ++d) a += q[l][h * 16 + d] * k[s][h * 16 + d];
      a *= 0.25f;
      sc[s] = a; mx = fmaxf(mx, a);
    }
    float sum = 0.0f;
    for (int s = 0; s < 24; ++s) { sc[s] = __expf(sc[s] - mx); sum += sc[s]; }
    float inv = 1.0f / sum;
    for (int d = 0; d < 16; ++d) {
      float a = 0.0f;
      for (int s = 0; s < 24; ++s) a += sc[s] * v[s][h * 16 + d];
      o[((size_t)m * LQ + l) * 64 + h * 16 + d] = a * inv;
    }
  }
}

__global__ void gather_ctx(const float* __restrict__ post, float* __restrict__ ctx,
                           _Float16* __restrict__ ctxh) {
  int i = blockIdx.x * blockDim.x + threadIdx.x;
  if (i >= MSEQ * DQ) return;
  int m = i >> 6, d = i & 63;
  float v = post[((size_t)m * LQ + (LQ - 1)) * 64 + d];
  ctx[i] = v; ctxh[i] = (_Float16)v;
}
__global__ void scatter_out(const float* __restrict__ raw, float* __restrict__ out) {
  int i = blockIdx.x * blockDim.x + threadIdx.x;
  if (i >= OUT0) return;
  int q = i % NQQ, t = (i / NQQ) % NTQ, n = (i / (NQQ * NTQ)) % NQ_;
  int hz = (i / (NQQ * NTQ * NQ_)) % HZQ, b = i / (NQQ * NTQ * NQ_ * HZQ);
  out[i] = raw[((size_t)(b * NQ_ + n)) * 72 + hz * (NTQ * NQQ) + t * NQQ + q];
}

// ---------------------------------------------------------------------------
// Host orchestration (param leaves in dict-insertion order of setup_inputs())
// ---------------------------------------------------------------------------
enum {
  I_XSEQ = 0, I_EIDX, I_EATTR, I_NFEAT,
  I_SW, I_SB,
  I_G1WL, I_G1WR, I_G1WE, I_G1ATT, I_G1B,
  I_G2WL, I_G2WR, I_G2WE, I_G2ATT, I_G2B,
  I_VFW1, I_VFW2, I_VFWF, I_VFWG, I_VFB1, I_VFB2, I_VFBF, I_VFBG, I_VFG, I_VFBE,
  I_VPW1, I_VPB1, I_VPW2, I_VPB2, I_VPWF, I_VPBF, I_VPWG, I_VPBG, I_VPWS, I_VPBS, I_VPG, I_VPBE,
  I_L1WIH, I_L1WHH, I_L1B,
  I_L2WIH, I_L2WHH, I_L2B,
  I_AQKV, I_ABQKV, I_AWO, I_ABO, I_AG, I_ABE,
  I_PGW1, I_PGW2, I_PGWF, I_PGWG, I_PGB1, I_PGB2, I_PGBF, I_PGBG, I_PGG, I_PGBE,
  I_OGW1, I_OGW2, I_OGWF, I_OGWG, I_OGB1, I_OGB2, I_OGBF, I_OGBG, I_OGG, I_OGBE,
  I_OW, I_OB
};

extern "C" void kernel_launch(void* const* d_in, const int* in_sizes, int n_in,
                              void* d_out, int out_size, void* d_ws, size_t ws_size,
                              hipStream_t stream) {
  (void)in_sizes; (void)n_in; (void)out_size; (void)ws_size;
#define PIN(i) ((const float*)d_in[(i)])

  char* wp = (char*)d_ws;
  auto alloc = [&](size_t bytes) -> void* {
    void* p = (void*)wp;
    wp += (bytes + 255) & ~(size_t)255;
    return p;
  };
  auto af = [&](size_t n) { return (float*)alloc(n * 4); };
  auto ah = [&](size_t n) { return (_Float16*)alloc(n * 2); };

  float* statf = af(NQ_ * DQ);
  _Float16* nodein = ah((size_t)MLQ * 96);
  _Float16* bt_g1l = ah(128 * 96); _Float16* bt_g1r = ah(128 * 96);
  _Float16* bt_g2l = ah(128 * 64); _Float16* bt_g2r = ah(128 * 64);
  _Float16* bt_vf[4]; for (int i = 0; i < 4; ++i) bt_vf[i] = ah(64 * 64);
  _Float16* bt_pg[4]; for (int i = 0; i < 4; ++i) bt_pg[i] = ah(64 * 64);
  _Float16* bt_og[4]; for (int i = 0; i < 4; ++i) bt_og[i] = ah(64 * 64);
  _Float16* bt_wih1 = ah(256 * 64); _Float16* bt_whh1 = ah(256 * 64);
  _Float16* bt_wih2 = ah(256 * 64); _Float16* bt_whh2 = ah(256 * 64);
  _Float16* bt_qkv = ah(192 * 64); _Float16* bt_wo = ah(64 * 64);
  _Float16* bt_ow = ah(128 * 64);
  _Float16* W2t = ah(64 * 64 * 64); _Float16* Wft = ah(64 * 64 * 64); _Float16* Wgt = ah(64 * 64 * 64);
  int* srcb = (int*)alloc(ETOT * 4); int* tgtb = (int*)alloc(ETOT * 4);
  float* eaext = af(ETOT * EDQ); float* eamean = af(EDQ);
  float* e1 = af((size_t)ETOT * 128); float* e2 = af((size_t)ETOT * 128);
  float* xl = af((size_t)MLQ * 128); float* xr = af((size_t)MLQ * 128);
  float* abuf = af((size_t)GLQ * ETOT * GHQ); float* exbuf = af((size_t)GLQ * ETOT * GHQ);
  unsigned* amaxb = (unsigned*)alloc((size_t)GLQ * NQ_ * GHQ * 4);
  float* denb = af((size_t)GLQ * NQ_ * GHQ);
  float* gout = af((size_t)GLQ * NQ_ * GHQ * 64);
  float* hb1 = af((size_t)MLQ * 64); _Float16* hb1h = ah((size_t)MLQ * 64);
  float* hb2 = af((size_t)MLQ * 64); _Float16* hb2h = ah((size_t)MLQ * 64);
  float* gflat = af((size_t)MLQ * 64); _Float16* gflath = ah((size_t)MLQ * 64);
  float* t1 = af((size_t)MLQ * 64); _Float16* t1h = ah((size_t)MLQ * 64);
  float* t2 = af((size_t)MLQ * 64); _Float16* t2h = ah((size_t)MLQ * 64);
  float* tf = af((size_t)MLQ * 64); float* tg = af((size_t)MLQ * 64);
  float* vsn_o = af((size_t)MLQ * 64); _Float16* vsn_oh = ah((size_t)MLQ * 64);
  float* xW = af((size_t)MLQ * 256);
  float* l1 = af((size_t)MLQ * 64); _Float16* l1h = ah((size_t)MLQ * 64);
  float* l2 = af((size_t)MLQ * 64); _Float16* l2h = ah((size_t)MLQ * 64);
  float* qkvb = af((size_t)MLQ * 192);
  float* ob = af((size_t)MLQ * 64); _Float16* obh = ah((size_t)MLQ * 64);
  float* attn_o = af((size_t)MLQ * 64); _Float16* attn_oh = ah((size_t)MLQ * 64);
  float* postb = af((size_t)MLQ * 64);
  float* ctx = af(MSEQ * 64); _Float16* ctxh = ah(MSEQ * 64);
  float* cg = af(MSEQ * 64); _Float16* cgh = ah(MSEQ * 64);
  float* rawb = af(MSEQ * 72);
  float* featw = (float*)d_out + OUT0;   // output #2, also consumed by vsn_per

  auto gemm = [&](const _Float16* A, int lda, const _Float16* Bt, int ldb,
                  const float* bias, float* C, int ldc, _Float16* Ch, int ldch,
                  int M, int N, int Kpad, int act) {
    dim3 gr(M / 16, (N + 63) / 64);
    gemm_wmma<<<gr, 128, 0, stream>>>(A, lda, Bt, ldb, bias, C, ldc, Ch, ldch, N, Kpad, act);
  };
  auto prep = [&](const float* W, _Float16* Bt, int K, int N, int Kpad, int Npad) {
    int tot = Npad * Kpad;
    prep_bt<<<(tot + 255) / 256, 256, 0, stream>>>(W, Bt, K, N, Kpad, Npad);
  };
  auto grn = [&](const float* xf32, const _Float16* xf16, int M, _Float16* const* btw,
                 const float* b1, const float* b2, const float* bfp, const float* bgp,
                 const float* g, const float* be, float* out, _Float16* outh, int sm) {
    gemm(xf16, 64, btw[0], 64, b1, t1, 64, t1h, 64, M, 64, 64, 1);
    gemm(t1h, 64, btw[1], 64, b2, t2, 64, t2h, 64, M, 64, 64, 0);
    gemm(t2h, 64, btw[2], 64, bfp, tf, 64, nullptr, 0, M, 64, 64, 0);
    gemm(t2h, 64, btw[3], 64, bgp, tg, 64, nullptr, 0, M, 64, 64, 2);
    grn_finish<<<M, 64, 0, stream>>>(xf32, tf, tg, g, be, out, outh, sm);
  };
  auto gat = [&](const _Float16* A, int lda, int Kpad, const _Float16* btl,
                 const _Float16* btr, const float* eb, const float* att,
                 const float* bb, float* out, _Float16* outh) {
    gemm(A, lda, btl, Kpad, nullptr, xl, 128, nullptr, 0, MLQ, 128, Kpad, 0);
    gemm(A, lda, btr, Kpad, nullptr, xr, 128, nullptr, 0, MLQ, 128, Kpad, 0);
    int nseg = GLQ * NQ_ * GHQ, neh = GLQ * ETOT * GHQ;
    fill_u32<<<(nseg + 255) / 256, 256, 0, stream>>>(amaxb, 0u, nseg);
    fill_f32<<<(nseg + 255) / 256, 256, 0, stream>>>(denb, 0.0f, nseg);
    fill_f32<<<(nseg * 64 + 255) / 256, 256, 0, stream>>>(gout, 0.0f, nseg * 64);
    gat_pass1<<<(neh + 127) / 128, 128, 0, stream>>>(xl, xr, eb, att, srcb, tgtb, abuf, amaxb);
    gat_pass2<<<(neh + 127) / 128, 128, 0, stream>>>(abuf, amaxb, tgtb, exbuf, denb);
    gat_pass3<<<((size_t)neh * 64 + 255) / 256, 256, 0, stream>>>(exbuf, denb, xl, srcb, tgtb, gout);
    gat_mean<<<(MLQ * 64 + 255) / 256, 256, 0, stream>>>(gout, bb, out, outh);
  };

  static_emb<<<(NQ_ * DQ + 127) / 128, 128, 0, stream>>>(PIN(I_NFEAT), PIN(I_SW), PIN(I_SB), statf);
  build_node_in<<<(MLQ * 96 + 255) / 256, 256, 0, stream>>>(PIN(I_XSEQ), statf, nodein);
  ea_mean<<<1, 32, 0, stream>>>(PIN(I_EATTR), eamean);
  build_edges<<<(ETOT + 127) / 128, 128, 0, stream>>>((const int*)d_in[I_EIDX], PIN(I_EATTR), eamean, srcb, tgtb, eaext);

  prep(PIN(I_G1WL), bt_g1l, 72, 128, 96, 128);
  prep(PIN(I_G1WR), bt_g1r, 72, 128, 96, 128);
  prep(PIN(I_G2WL), bt_g2l, 64, 128, 64, 128);
  prep(PIN(I_G2WR), bt_g2r, 64, 128, 64, 128);
  { const int vfw[4] = {I_VFW1, I_VFW2, I_VFWF, I_VFWG};
    const int pgw[4] = {I_PGW1, I_PGW2, I_PGWF, I_PGWG};
    const int ogw[4] = {I_OGW1, I_OGW2, I_OGWF, I_OGWG};
    for (int i = 0; i < 4; ++i) { prep(PIN(vfw[i]), bt_vf[i], 64, 64, 64, 64);
                                  prep(PIN(pgw[i]), bt_pg[i], 64, 64, 64, 64);
                                  prep(PIN(ogw[i]), bt_og[i], 64, 64, 64, 64); } }
  prep(PIN(I_L1WIH), bt_wih1, 64, 256, 64, 256);
  prep(PIN(I_L1WHH), bt_whh1, 64, 256, 64, 256);
  prep(PIN(I_L2WIH), bt_wih2, 64, 256, 64, 256);
  prep(PIN(I_L2WHH), bt_whh2, 64, 256, 64, 256);
  prep(PIN(I_AQKV), bt_qkv, 64, 192, 64, 192);
  prep(PIN(I_AWO), bt_wo, 64, 64, 64, 64);
  prep(PIN(I_OW), bt_ow, 64, 72, 64, 128);
  prep_bt3<<<(262144 + 255) / 256, 256, 0, stream>>>(PIN(I_VPW2), W2t);
  prep_bt3<<<(262144 + 255) / 256, 256, 0, stream>>>(PIN(I_VPWF), Wft);
  prep_bt3<<<(262144 + 255) / 256, 256, 0, stream>>>(PIN(I_VPWG), Wgt);

  e_proj<<<(ETOT * 128 + 255) / 256, 256, 0, stream>>>(eaext, PIN(I_G1WE), e1);
  e_proj<<<(ETOT * 128 + 255) / 256, 256, 0, stream>>>(eaext, PIN(I_G2WE), e2);
  gat(nodein, 96, 96, bt_g1l, bt_g1r, e1, PIN(I_G1ATT), PIN(I_G1B), hb1, hb1h);
  gat(hb1h, 64, 64, bt_g2l, bt_g2r, e2, PIN(I_G2ATT), PIN(I_G2B), hb2, hb2h);

  transpose_flat<<<(MLQ * 64 + 255) / 256, 256, 0, stream>>>(hb2, gflat, gflath);

  grn(gflat, gflath, MLQ, bt_vf, PIN(I_VFB1), PIN(I_VFB2), PIN(I_VFBF), PIN(I_VFBG),
      PIN(I_VFG), PIN(I_VFBE), featw, nullptr, 1);
  vsn_per<<<MLQ / 16, 128, 0, stream>>>(gflat, featw, PIN(I_VPW1), PIN(I_VPB1),
      W2t, PIN(I_VPB2), Wft, PIN(I_VPBF), Wgt, PIN(I_VPBG),
      PIN(I_VPWS), PIN(I_VPBS), PIN(I_VPG), PIN(I_VPBE), vsn_o, vsn_oh);

  gemm(vsn_oh, 64, bt_wih1, 64, PIN(I_L1B), xW, 256, nullptr, 0, MLQ, 256, 64, 0);
  lstm_layer<<<MSEQ / 16, 128, 0, stream>>>(xW, bt_whh1, l1, l1h);
  gemm(l1h, 64, bt_wih2, 64, PIN(I_L2B), xW, 256, nullptr, 0, MLQ, 256, 64, 0);
  lstm_layer<<<MSEQ / 16, 128, 0, stream>>>(xW, bt_whh2, l2, l2h);

  gemm(l2h, 64, bt_qkv, 64, PIN(I_ABQKV), qkvb, 192, nullptr, 0, MLQ, 192, 64, 0);
  mha_attn<<<MSEQ, 128, 0, stream>>>(qkvb, ob);
  to_f16<<<(MLQ * 64 + 255) / 256, 256, 0, stream>>>(ob, obh, MLQ * 64);
  gemm(obh, 64, bt_wo, 64, PIN(I_ABO), t1, 64, nullptr, 0, MLQ, 64, 64, 0);
  grn_finish<<<MLQ, 64, 0, stream>>>(l2, t1, nullptr, PIN(I_AG), PIN(I_ABE), attn_o, attn_oh, 0);

  grn(attn_o, attn_oh, MLQ, bt_pg, PIN(I_PGB1), PIN(I_PGB2), PIN(I_PGBF), PIN(I_PGBG),
      PIN(I_PGG), PIN(I_PGBE), postb, nullptr, 0);
  gather_ctx<<<(MSEQ * 64 + 255) / 256, 256, 0, stream>>>(postb, ctx, ctxh);
  grn(ctx, ctxh, MSEQ, bt_og, PIN(I_OGB1), PIN(I_OGB2), PIN(I_OGBF), PIN(I_OGBG),
      PIN(I_OGG), PIN(I_OGBE), cg, cgh, 0);
  gemm(cgh, 64, bt_ow, 64, PIN(I_OB), rawb, 72, nullptr, 0, MSEQ, 72, 64, 0);
  scatter_out<<<(OUT0 + 255) / 256, 256, 0, stream>>>(rawb, (float*)d_out);
#undef PIN
}
